// USA_61143154426146
// MI455X (gfx1250) — matrix operand order repaired and checked by reading, not compiled, not run on hardware
//
#include <hip/hip_runtime.h>

typedef __attribute__((ext_vector_type(16))) _Float16 v16h;
typedef __attribute__((ext_vector_type(8)))  _Float16 v8h;
typedef __attribute__((ext_vector_type(8)))  float    v8f;
typedef __attribute__((ext_vector_type(4)))  float    v4f;

#define HH   32
#define SS   2048
#define DD   128
#define II   128
#define FF   32
#define LDSP 136   // padded LDS row stride in halves (breaks 64-bank conflicts)

// ---------------------------------------------------------------------------
// Fragment loaders (CDNA5 wave32 WMMA layouts, CDNA5 ISA §7.12.2)
// A (16x32 f16): lanes 0-15 -> row M=lane, elems 0..7 = K0..7, 8..15 = K16..23
//                lanes 16-31 -> row M=lane-16, elems 0..7 = K8..15, 8..15 = K24..31
// ---------------------------------------------------------------------------
__device__ inline v16h load_a_f32(const float* __restrict__ base, int ld, int k0, int lane) {
    int row  = lane & 15;
    int koff = (lane >> 4) << 3;             // 0 or 8
    const float* p0 = base + (size_t)row * ld + k0 + koff;
    const float* p1 = p0 + 16;
    v16h a;
#pragma unroll
    for (int i = 0; i < 8; ++i) {
        a[i]     = (_Float16)p0[i];
        a[i + 8] = (_Float16)p1[i];
    }
    return a;
}

__device__ inline v16h load_a_h(const _Float16* __restrict__ base, int ld, int k0, int lane) {
    int row  = lane & 15;
    int koff = (lane >> 4) << 3;
    const _Float16* p0 = base + (size_t)row * ld + k0 + koff;
    v8h lo = *(const v8h*)p0;
    v8h hi = *(const v8h*)(p0 + 16);
    v16h a;
#pragma unroll
    for (int i = 0; i < 8; ++i) { a[i] = lo[i]; a[i + 8] = hi[i]; }
    return a;
}

// B (32x16 f16): lanes 0-15 -> col N=lane, elems 0..7 = K0..7, 8..15 = K16..23; lanes 16-31 -> col N=lane-16, elems 0..7 = K8..15, 8..15 = K24..31
// Weights are [out,in] row-major, y = x @ W^T, so B column n == W row n -> contiguous.
__device__ inline v16h load_b_w(const _Float16* __restrict__ w, int ld, int n0, int k0, int lane) {
    const _Float16* p = w + (size_t)(n0 + (lane & 15)) * ld + k0 + ((lane >> 4) << 3);
    v8h lo = *(const v8h*)p;
    v8h hi = *(const v8h*)(p + 16);
    v16h b;
#pragma unroll
    for (int i = 0; i < 8; ++i) { b[i] = lo[i]; b[i + 8] = hi[i]; }
    return b;
}

__device__ inline v8f wmma_f16(v16h a, v16h b, v8f c) {
    return __builtin_amdgcn_wmma_f32_16x16x32_f16(false, a, false, b, (short)0, c, false, false);
}

__device__ inline float silu_f(float x) { return x / (1.0f + __expf(-x)); }

// ---------------------------------------------------------------------------
// f32 -> f16 weight conversion
// ---------------------------------------------------------------------------
__global__ void usa_cvt_h(const float* __restrict__ src, _Float16* __restrict__ dst, int n) {
    int i = blockIdx.x * blockDim.x + threadIdx.x;
    if (i < n) dst[i] = (_Float16)src[i];
}

// ---------------------------------------------------------------------------
// Stage 1: per-head 3-layer MLP + tanh.  X:[H,S,D] f32 -> Out:[H,S,F] f16
// One wave per 16-row tile; 4 waves per block. Accumulate from inline-zero C;
// bias is per-column, folded into the activation epilogue (no C broadcast movs).
// ---------------------------------------------------------------------------
__global__ __launch_bounds__(128) void usa_mlp_kernel(
    const float*    __restrict__ X,
    const _Float16* __restrict__ W1h,  // [H, II, DD]
    const float*    __restrict__ b1,   // [H, II]
    const _Float16* __restrict__ W2h,  // [H, II, II]
    const float*    __restrict__ b2,   // [H, II]
    const _Float16* __restrict__ W3h,  // [H, FF, II]
    const float*    __restrict__ b3,   // [H, FF]
    _Float16*       __restrict__ Out)  // [H, S, FF]
{
    __shared__ _Float16 lds[4][16][LDSP];
    const int wave = threadIdx.x >> 5;
    const int lane = threadIdx.x & 31;
    const int gtile = blockIdx.x * 4 + wave;     // 0 .. H*(S/16)-1
    const int h  = gtile >> 7;                   // 128 row-tiles per head
    const int rt = gtile & 127;

    const float*    xb = X   + ((size_t)h * SS + rt * 16) * DD;
    const _Float16* w1 = W1h + (size_t)h * II * DD;
    const _Float16* w2 = W2h + (size_t)h * II * II;
    const _Float16* w3 = W3h + (size_t)h * FF * II;
    _Float16 (*sm)[LDSP] = lds[wave];

    const int col   = lane & 15;
    const int rowhi = (lane >> 4) << 3;          // 0 or 8: C/D row offset

    // ---- layer 1: [16,128] @ W1^T -> silu -> LDS [16,128]
    v16h a0 = load_a_f32(xb, DD, 0,  lane);
    v16h a1 = load_a_f32(xb, DD, 32, lane);
    v16h a2 = load_a_f32(xb, DD, 64, lane);
    v16h a3 = load_a_f32(xb, DD, 96, lane);
#pragma unroll
    for (int nt = 0; nt < 8; ++nt) {
        v8f c = {};
        c = wmma_f16(a0, load_b_w(w1, DD, nt * 16, 0,  lane), c);
        c = wmma_f16(a1, load_b_w(w1, DD, nt * 16, 32, lane), c);
        c = wmma_f16(a2, load_b_w(w1, DD, nt * 16, 64, lane), c);
        c = wmma_f16(a3, load_b_w(w1, DD, nt * 16, 96, lane), c);
        float bv = b1[h * II + nt * 16 + col];
#pragma unroll
        for (int v = 0; v < 8; ++v)
            sm[v + rowhi][nt * 16 + col] = (_Float16)silu_f(c[v] + bv);
    }
    asm volatile("s_wait_dscnt 0" ::: "memory");

    // ---- layer 2: LDS [16,128] @ W2^T -> silu -> LDS
    a0 = load_a_h(&sm[0][0], LDSP, 0,  lane);
    a1 = load_a_h(&sm[0][0], LDSP, 32, lane);
    a2 = load_a_h(&sm[0][0], LDSP, 64, lane);
    a3 = load_a_h(&sm[0][0], LDSP, 96, lane);
#pragma unroll
    for (int nt = 0; nt < 8; ++nt) {
        v8f c = {};
        c = wmma_f16(a0, load_b_w(w2, II, nt * 16, 0,  lane), c);
        c = wmma_f16(a1, load_b_w(w2, II, nt * 16, 32, lane), c);
        c = wmma_f16(a2, load_b_w(w2, II, nt * 16, 64, lane), c);
        c = wmma_f16(a3, load_b_w(w2, II, nt * 16, 96, lane), c);
        float bv = b2[h * II + nt * 16 + col];
#pragma unroll
        for (int v = 0; v < 8; ++v)
            sm[v + rowhi][nt * 16 + col] = (_Float16)silu_f(c[v] + bv);
    }
    asm volatile("s_wait_dscnt 0" ::: "memory");

    // ---- layer 3: LDS [16,128] @ W3^T (F=32) -> tanh -> global f16
    a0 = load_a_h(&sm[0][0], LDSP, 0,  lane);
    a1 = load_a_h(&sm[0][0], LDSP, 32, lane);
    a2 = load_a_h(&sm[0][0], LDSP, 64, lane);
    a3 = load_a_h(&sm[0][0], LDSP, 96, lane);
    _Float16* ob = Out + ((size_t)h * SS + rt * 16) * FF;
#pragma unroll
    for (int nt = 0; nt < 2; ++nt) {
        v8f c = {};
        c = wmma_f16(a0, load_b_w(w3, II, nt * 16, 0,  lane), c);
        c = wmma_f16(a1, load_b_w(w3, II, nt * 16, 32, lane), c);
        c = wmma_f16(a2, load_b_w(w3, II, nt * 16, 64, lane), c);
        c = wmma_f16(a3, load_b_w(w3, II, nt * 16, 96, lane), c);
        float bv = b3[h * FF + nt * 16 + col];
#pragma unroll
        for (int v = 0; v < 8; ++v)
            ob[(size_t)(v + rowhi) * FF + nt * 16 + col] = (_Float16)tanhf(c[v] + bv);
    }
}

// ---------------------------------------------------------------------------
// Stage 2: scores = sigmoid(Ql @ Kl^T - 0.5) with causal mask.
// One wave per 16x32 output tile: A fragment reused across two WMMAs
// (K = F = 32 -> exactly one WMMA per 16x16 subtile).  Fully-masked subtiles
// are zero-filled with b128 stores.  Output-store bound: 537 MB -> ~23 us
// floor at 23.3 TB/s HBM.
// ---------------------------------------------------------------------------
__global__ __launch_bounds__(256) void usa_scores_kernel(
    const _Float16* __restrict__ Ql,  // [H,S,F]
    const _Float16* __restrict__ Kl,  // [H,S,F]
    float*          __restrict__ Out) // [H,S,S]
{
    const int wave = threadIdx.x >> 5;
    const int lane = threadIdx.x & 31;
    const long tile = (long)blockIdx.x * 8 + wave;   // H * 128 * 64 tiles
    const int h   = (int)(tile >> 13);               // 128*64 = 8192 tiles/head
    const int rem = (int)(tile & 8191);
    const int sq0 = (rem >> 6) << 4;                 // query-row base
    const int sk0 = (rem & 63) << 5;                 // key-col base (32 wide)

    const int col   = lane & 15;
    const int rowhi = (lane >> 4) << 3;
    float* ob = Out + ((size_t)h * SS + sq0) * SS + sk0;

    // A: 16 query rows x 32 features (shared by both subtiles)
    const _Float16* qb = Ql + ((size_t)h * SS + sq0) * FF;
    v16h a = load_a_h(qb, FF, 0, lane);

    const _Float16* kbase = Kl + (size_t)h * SS * FF;

    // zero-fill lane mapping: 2 x (8 rows x 16 cols) via b128 stores
    const int zrow = lane >> 2;          // 0..7
    const int zcol = (lane & 3) << 2;    // 0,4,8,12

#pragma unroll
    for (int half = 0; half < 2; ++half) {
        const int sk = sk0 + half * 16;
        float* obh = ob + half * 16;

        if (sk > sq0 + 15) {
            // fully above diagonal: sigmoid(-FLT_MAX - 0.5) == 0
            v4f z = {};
            *(v4f*)(obh + (size_t)zrow       * SS + zcol) = z;
            *(v4f*)(obh + (size_t)(zrow + 8) * SS + zcol) = z;
            continue;
        }

        // B: Kl^T -> column n is Kl row (sk+n); two 8-half chunks per lane (K 8h..8h+7, 16+8h..)
        const _Float16* kb = kbase + (size_t)(sk + col) * FF + ((lane >> 4) << 3);
        v8h lo = *(const v8h*)kb;
        v8h hi = *(const v8h*)(kb + 16);
        v16h b;
#pragma unroll
        for (int i = 0; i < 8; ++i) { b[i] = lo[i]; b[i + 8] = hi[i]; }

        v8f c = {};
        c = wmma_f16(a, b, c);

#pragma unroll
        for (int v = 0; v < 8; ++v) {
            int t = sq0 + v + rowhi;
            int s = sk + col;
            float val = (s <= t) ? (1.0f / (1.0f + __expf(0.5f - c[v]))) : 0.0f;
            obh[(size_t)(v + rowhi) * SS + col] = val;
        }
    }
}

// ---------------------------------------------------------------------------
extern "C" void kernel_launch(void* const* d_in, const int* in_sizes, int n_in,
                              void* d_out, int out_size, void* d_ws, size_t ws_size,
                              hipStream_t stream) {
    const float* K   = (const float*)d_in[0];
    const float* Q   = (const float*)d_in[1];
    const float* Wk1 = (const float*)d_in[2];
    const float* bk1 = (const float*)d_in[3];
    const float* Wk2 = (const float*)d_in[4];
    const float* bk2 = (const float*)d_in[5];
    const float* Wk3 = (const float*)d_in[6];
    const float* bk3 = (const float*)d_in[7];
    const float* Wq1 = (const float*)d_in[8];
    const float* bq1 = (const float*)d_in[9];
    const float* Wq2 = (const float*)d_in[10];
    const float* bq2 = (const float*)d_in[11];
    const float* Wq3 = (const float*)d_in[12];
    const float* bq3 = (const float*)d_in[13];
    float* out = (float*)d_out;

    // carve f16 workspace (256B aligned chunks)
    char* ws = (char*)d_ws;
    size_t off = 0;
    auto carve = [&](size_t elems) -> _Float16* {
        _Float16* p = (_Float16*)(ws + off);
        off = (off + elems * sizeof(_Float16) + 255) & ~(size_t)255;
        return p;
    };
    const int nW1 = HH * II * DD, nW2 = HH * II * II, nW3 = HH * FF * II;
    _Float16* Qlh  = carve((size_t)HH * SS * FF);
    _Float16* Klh  = carve((size_t)HH * SS * FF);
    _Float16* Wk1h = carve(nW1);
    _Float16* Wk2h = carve(nW2);
    _Float16* Wk3h = carve(nW3);
    _Float16* Wq1h = carve(nW1);
    _Float16* Wq2h = carve(nW2);
    _Float16* Wq3h = carve(nW3);

    // weight conversion f32 -> f16
    usa_cvt_h<<<(nW1 + 255) / 256, 256, 0, stream>>>(Wk1, Wk1h, nW1);
    usa_cvt_h<<<(nW2 + 255) / 256, 256, 0, stream>>>(Wk2, Wk2h, nW2);
    usa_cvt_h<<<(nW3 + 255) / 256, 256, 0, stream>>>(Wk3, Wk3h, nW3);
    usa_cvt_h<<<(nW1 + 255) / 256, 256, 0, stream>>>(Wq1, Wq1h, nW1);
    usa_cvt_h<<<(nW2 + 255) / 256, 256, 0, stream>>>(Wq2, Wq2h, nW2);
    usa_cvt_h<<<(nW3 + 255) / 256, 256, 0, stream>>>(Wq3, Wq3h, nW3);

    // stage 1: MLP+tanh for K and Q  (H*S/16 = 4096 waves, 4 waves/block)
    usa_mlp_kernel<<<1024, 128, 0, stream>>>(K, Wk1h, bk1, Wk2h, bk2, Wk3h, bk3, Klh);
    usa_mlp_kernel<<<1024, 128, 0, stream>>>(Q, Wq1h, bq1, Wq2h, bq2, Wq3h, bq3, Qlh);

    // stage 2: masked sigmoid scores (H*128*64 = 262144 tiles, 8 waves/block)
    usa_scores_kernel<<<32768, 256, 0, stream>>>(Qlh, Klh, out);
}
